// Head_25443386261742
// MI455X (gfx1250) — matrix operand order repaired and checked by reading, not compiled, not run on hardware
//
#include <hip/hip_runtime.h>

#define EMB  512
#define HEAD 64
#define BATCH 4
#define SEQ  4096

typedef __attribute__((ext_vector_type(16))) _Float16 v16h;
typedef __attribute__((ext_vector_type(8)))  float    v8f;
typedef int v4i_t __attribute__((vector_size(16)));  // async-LDS builtin ptr type

#if defined(__has_builtin)
#if __has_builtin(__builtin_amdgcn_global_load_async_to_lds_b128) && \
    __has_builtin(__builtin_amdgcn_s_wait_asynccnt)
#define USE_ASYNC_LDS 1
#endif
#endif

static __device__ __forceinline__ v8f wmma_f16(v16h a, v16h b, v8f c) {
  // 8 args: (neg_a, A, neg_b, B, c_mod, C, reuse_a, reuse_b)
  return __builtin_amdgcn_wmma_f32_16x16x32_f16(false, a, false, b, (short)0, c,
                                                false, false);
}

// ---------------------------------------------------------------------------
// QKV projection: [B*S, EMB] x [EMB, HEAD] -> f16 [B*S, HEAD]
// One wave computes a 16-row x 64-col strip with V_WMMA_F32_16X16X32_F16.
// blockIdx.x = 16-row tile over B*S, blockIdx.y = which projection (Q/K/V)
// ---------------------------------------------------------------------------
__global__ __launch_bounds__(32) void proj_wmma(
    const float* __restrict__ xq, const float* __restrict__ xk,
    const float* __restrict__ xv, const float* __restrict__ Wq,
    const float* __restrict__ Wk, const float* __restrict__ Wv,
    _Float16* __restrict__ Qh, _Float16* __restrict__ Kh,
    _Float16* __restrict__ Vh) {
  const int tile  = blockIdx.x;
  const int which = blockIdx.y;
  const float* X   = (which == 0) ? xq : (which == 1) ? xk : xv;
  const float* W   = (which == 0) ? Wq : (which == 1) ? Wk : Wv;
  _Float16*    Out = (which == 0) ? Qh : (which == 1) ? Kh : Vh;

  const int  lane = threadIdx.x & 31;
  const int  m    = lane & 15;   // A-row / B-col / D-col index
  const int  hi   = lane >> 4;   // lane half
  const long row0 = (long)tile * 16;

  v8f acc[4] = {};
  for (int kc = 0; kc < EMB / 32; ++kc) {
    // A operand: 16x32 f16, ISA layout (lanes 0-15: K 0..7,16..23; 16-31: +8)
    const float* arow = X + (row0 + m) * EMB + kc * 32;
    v16h a;
#pragma unroll
    for (int i = 0; i < 8; ++i) {
      const int kk = ((i >> 2) << 4) + (hi << 3) + ((i & 3) << 1);
      a[2 * i]     = (_Float16)arow[kk];
      a[2 * i + 1] = (_Float16)arow[kk + 1];
    }
#pragma unroll
    for (int nt = 0; nt < 4; ++nt) {
      // B operand: 32x16, B[kk][n] = W[n][kk]; lanes 0-15 K 0..7,16..23; 16-31: +8
      const float* wrow = W + (nt * 16 + m) * EMB + kc * 32;
      v16h b;
#pragma unroll
      for (int i = 0; i < 8; ++i) {
        const int kk = ((i >> 2) << 4) + (hi << 3) + ((i & 3) << 1);
        b[2 * i]     = (_Float16)wrow[kk];
        b[2 * i + 1] = (_Float16)wrow[kk + 1];
      }
      acc[nt] = wmma_f16(a, b, acc[nt]);
    }
  }
  // D layout: VGPR j -> row j + 8*hi, lane -> col (lane & 15)
#pragma unroll
  for (int nt = 0; nt < 4; ++nt)
#pragma unroll
    for (int j = 0; j < 8; ++j)
      Out[(row0 + j + 8 * hi) * HEAD + nt * 16 + m] = (_Float16)acc[nt][j];
}

// ---------------------------------------------------------------------------
// Flash attention: each wave owns a 16-row Q tile; 4 waves per workgroup share
// 32-key K/V blocks staged in LDS via GLOBAL_LOAD_ASYNC_TO_LDS_B128 (ASYNCcnt,
// s_wait_asynccnt). Online softmax in registers; P tile bounced through
// per-wave LDS to convert D-layout -> A-layout (DS ops are in-order per wave).
// ---------------------------------------------------------------------------
__global__ __launch_bounds__(128) void flash_attn_wmma(
    const _Float16* __restrict__ Qh, const _Float16* __restrict__ Kh,
    const _Float16* __restrict__ Vh, float* __restrict__ out) {
  __shared__ _Float16 ldsK[32 * 64];      // [key][emb]
  __shared__ _Float16 ldsV[32 * 64];      // [key][head] (row major)
  __shared__ _Float16 ldsP[4 * 16 * 32];  // per-wave [row][col]

  const int  tid  = threadIdx.x;
  const int  lane = tid & 31;
  const int  wave = tid >> 5;
  const int  m    = lane & 15;
  const int  hi   = lane >> 4;
  const int  b    = blockIdx.y;
  const long qrow0 = (long)blockIdx.x * 64 + wave * 16;  // row within batch

  const _Float16* Qb = Qh + (long)b * SEQ * HEAD;
  const _Float16* Kb = Kh + (long)b * SEQ * HEAD;
  const _Float16* Vb = Vh + (long)b * SEQ * HEAD;

  // Q tile A-operands, loaded once, live in VGPRs across the whole key loop
  v16h qa[2];
  {
    const _Float16* qr = Qb + (qrow0 + m) * HEAD;
#pragma unroll
    for (int c = 0; c < 2; ++c)
#pragma unroll
      for (int i = 0; i < 8; ++i) {
        const int kk = c * 32 + ((i >> 2) << 4) + (hi << 3) + ((i & 3) << 1);
        qa[c][2 * i]     = qr[kk];
        qa[c][2 * i + 1] = qr[kk + 1];
      }
  }

  float mrow[8], lrow[8];
  v8f   o[4] = {};
#pragma unroll
  for (int j = 0; j < 8; ++j) { mrow[j] = -1e30f; lrow[j] = 0.0f; }

  const float scale = 0.125f;  // 1/sqrt(HEAD)
  const int   key  = tid >> 2; // staging: key row 0..31
  const int   part = tid & 3;  // 16-half segment of a 64-half row

  for (int kb = 0; kb < SEQ / 32; ++kb) {
    __syncthreads();  // everyone done reading previous K/V block
    // ---- stage K and V blocks [32 x 64] into LDS ----
    {
      const _Float16* ksrc = Kb + ((long)kb * 32 + key) * HEAD + part * 16;
      const _Float16* vsrc = Vb + ((long)kb * 32 + key) * HEAD + part * 16;
      _Float16* kdst = &ldsK[key * 64 + part * 16];
      _Float16* vdst = &ldsV[key * 64 + part * 16];
#if defined(USE_ASYNC_LDS)
      // 16B per lane per op; offsets 0 and 16 cover the 32B segment.
      __builtin_amdgcn_global_load_async_to_lds_b128(
          (v4i_t*)ksrc, (v4i_t*)kdst, 0, 0);
      __builtin_amdgcn_global_load_async_to_lds_b128(
          (v4i_t*)ksrc, (v4i_t*)kdst, 16, 0);
      __builtin_amdgcn_global_load_async_to_lds_b128(
          (v4i_t*)vsrc, (v4i_t*)vdst, 0, 0);
      __builtin_amdgcn_global_load_async_to_lds_b128(
          (v4i_t*)vsrc, (v4i_t*)vdst, 16, 0);
#else
      const uint4* ks4 = reinterpret_cast<const uint4*>(ksrc);
      const uint4* vs4 = reinterpret_cast<const uint4*>(vsrc);
      uint4* kd4 = reinterpret_cast<uint4*>(kdst);
      uint4* vd4 = reinterpret_cast<uint4*>(vdst);
      kd4[0] = ks4[0];
      kd4[1] = ks4[1];
      vd4[0] = vs4[0];
      vd4[1] = vs4[1];
#endif
    }
    if (kb + 1 < SEQ / 32) {  // global_prefetch_b8 for next block
      __builtin_prefetch(Kb + ((long)(kb + 1) * 32 + key) * HEAD + part * 16, 0, 0);
      __builtin_prefetch(Vb + ((long)(kb + 1) * 32 + key) * HEAD + part * 16, 0, 0);
    }
#if defined(USE_ASYNC_LDS)
    __builtin_amdgcn_s_wait_asynccnt(0);
#endif
    __syncthreads();

    // ---- scores S = Q * K^T (16 x 32), two 16x16 D tiles ----
    v8f s0 = {}, s1 = {};
#pragma unroll
    for (int kc = 0; kc < 2; ++kc) {
      v16h b0, b1;
#pragma unroll
      for (int i = 0; i < 8; ++i) {
        const int kk = ((i >> 2) << 4) + (hi << 3) + ((i & 3) << 1);
        const _Float16* p0 = &ldsK[m * 64 + kc * 32 + kk];         // keys 0..15
        const _Float16* p1 = &ldsK[(16 + m) * 64 + kc * 32 + kk];  // keys 16..31
        b0[2 * i] = p0[0]; b0[2 * i + 1] = p0[1];
        b1[2 * i] = p1[0]; b1[2 * i + 1] = p1[1];
      }
      s0 = wmma_f16(qa[kc], b0, s0);
      s1 = wmma_f16(qa[kc], b1, s1);
    }

    // ---- online softmax over the 32 new columns ----
    float pm[8];
#pragma unroll
    for (int j = 0; j < 8; ++j) {
      s0[j] *= scale;
      s1[j] *= scale;
      pm[j] = fmaxf(s0[j], s1[j]);
    }
    // row lives across 16 lanes of one half: xor masks 1..8 stay in-half
#pragma unroll
    for (int mask = 1; mask <= 8; mask <<= 1)
#pragma unroll
      for (int j = 0; j < 8; ++j)
        pm[j] = fmaxf(pm[j], __shfl_xor(pm[j], mask, 32));

    float corr[8];
#pragma unroll
    for (int j = 0; j < 8; ++j) {
      const float mn = fmaxf(mrow[j], pm[j]);
      corr[j] = __expf(mrow[j] - mn);
      mrow[j] = mn;
    }
    float ls[8];
#pragma unroll
    for (int j = 0; j < 8; ++j) {
      const float p0 = __expf(s0[j] - mrow[j]);
      const float p1 = __expf(s1[j] - mrow[j]);
      ls[j] = p0 + p1;
      _Float16* pp = &ldsP[wave * 512 + (j + 8 * hi) * 32];
      pp[m]      = (_Float16)p0;   // D-layout -> LDS row-major
      pp[16 + m] = (_Float16)p1;
    }
#pragma unroll
    for (int mask = 1; mask <= 8; mask <<= 1)
#pragma unroll
      for (int j = 0; j < 8; ++j)
        ls[j] += __shfl_xor(ls[j], mask, 32);
#pragma unroll
    for (int j = 0; j < 8; ++j)
      lrow[j] = lrow[j] * corr[j] + ls[j];
#pragma unroll
    for (int nt = 0; nt < 4; ++nt)
#pragma unroll
      for (int j = 0; j < 8; ++j)
        o[nt][j] *= corr[j];

    // ---- O += P * V ----
    v16h pa;  // reload P in A-layout (same-wave DS ops are in-order)
#pragma unroll
    for (int i = 0; i < 8; ++i) {
      const int kk = ((i >> 2) << 4) + (hi << 3) + ((i & 3) << 1);
      const _Float16* p = &ldsP[wave * 512 + m * 32 + kk];
      pa[2 * i] = p[0];
      pa[2 * i + 1] = p[1];
    }
#pragma unroll
    for (int nt = 0; nt < 4; ++nt) {
      v16h bv;  // B[kk][n] = V[key kk][head nt*16+n] (row-major LDS)
#pragma unroll
      for (int i = 0; i < 8; ++i) {
        const int kk = ((i >> 2) << 4) + (hi << 3) + ((i & 3) << 1);
        bv[2 * i]     = ldsV[kk * 64 + nt * 16 + m];
        bv[2 * i + 1] = ldsV[(kk + 1) * 64 + nt * 16 + m];
      }
      o[nt] = wmma_f16(pa, bv, o[nt]);
    }
  }

  // ---- finalize: divide by row sums, store fp32 ----
#pragma unroll
  for (int nt = 0; nt < 4; ++nt)
#pragma unroll
    for (int j = 0; j < 8; ++j) {
      const long r = (long)b * SEQ + qrow0 + j + 8 * hi;
      out[r * HEAD + nt * 16 + m] = o[nt][j] / lrow[j];
    }
}

extern "C" void kernel_launch(void* const* d_in, const int* in_sizes, int n_in,
                              void* d_out, int out_size, void* d_ws,
                              size_t ws_size, hipStream_t stream) {
  (void)in_sizes; (void)n_in; (void)out_size; (void)ws_size;
  const float* q  = (const float*)d_in[0];
  const float* k  = (const float*)d_in[1];
  const float* v  = (const float*)d_in[2];
  const float* Wq = (const float*)d_in[3];
  const float* Wk = (const float*)d_in[4];
  const float* Wv = (const float*)d_in[5];
  float* outp = (float*)d_out;

  const size_t n = (size_t)BATCH * SEQ * HEAD;  // 2 MiB of halves each
  _Float16* Qh = (_Float16*)d_ws;
  _Float16* Kh = Qh + n;
  _Float16* Vh = Kh + n;

  // 1024 row-tiles x {Q,K,V}, one wave each
  proj_wmma<<<dim3((BATCH * SEQ) / 16, 3), 32, 0, stream>>>(q, k, v, Wq, Wk, Wv,
                                                            Qh, Kh, Vh);
  // 64 q-blocks of 64 rows x 4 batches, 4 waves per block
  flash_attn_wmma<<<dim3(SEQ / 64, BATCH), 128, 0, stream>>>(Qh, Kh, Vh, outp);
}
